// GATLayer_42588895707588
// MI455X (gfx1250) — matrix operand order repaired and checked by reading, not compiled, not run on hardware
//
#include <hip/hip_runtime.h>
#include <hip/hip_bf16.h>

#define N_NODES 4096
#define IN_FEAT 256
#define NHEAD   8
#define FDIM    64
#define HF      512     // NHEAD*FDIM
#define ALPHA   0.2f

typedef __attribute__((ext_vector_type(16))) __bf16 v16bf;
typedef __attribute__((ext_vector_type(8)))  __bf16 v8bf;
typedef __attribute__((ext_vector_type(2)))  __bf16 v2bf;
typedef __attribute__((ext_vector_type(8)))  float  v8f;

// packed f32x2 -> bf16x2. Prefer the explicit builtin when the toolchain has
// it; otherwise use native __bf16 fptrunc, which the gfx1250 backend can pair
// into v_cvt_pk_bf16_f32 (CDNA5 has hardware bf16 converts).
__device__ __forceinline__ v2bf pack2bf(float lo, float hi) {
#if __has_builtin(__builtin_amdgcn_cvt_pk_bf16_f32)
    return __builtin_amdgcn_cvt_pk_bf16_f32(lo, hi);
#else
    v2bf r;
    r[0] = (__bf16)lo;
    r[1] = (__bf16)hi;
    return r;
#endif
}

__device__ __forceinline__ unsigned short f2bfbits(float f) {
    union { __bf16 b; unsigned short s; } c;
    c.b = (__bf16)f;
    return c.s;
}

// build a 16-element bf16 fragment from 8 packed pairs (elements 2i,2i+1
// share a VGPR -> pairs map 1:1 onto the WMMA 16-bit operand layout)
union FragBF16 {
    v16bf v;
    v2bf  p[8];
    __device__ FragBF16() {}
};

// masked softmax probability for one edge
__device__ __forceinline__ float gat_p(int adjv, float s1, float a2v,
                                       float mi, float inv_l) {
    float x = s1 + a2v;
    x = (x > 0.f) ? x : ALPHA * x;           // leaky relu
    float p = __expf(x - mi) * inv_l;
    return (adjv > 0) ? p : 0.f;
}

// ---------------------------------------------------------------------------
// K1: Wh = h @ W  (bf16 WMMA). One wave per 16x16 output tile, K=256 in 8 steps.
// Also emits a head-major, f-major bf16 transpose Whbf[h][f][j] for pass 4.
// ---------------------------------------------------------------------------
__global__ __launch_bounds__(256) void k_gemm_wh(
    const float* __restrict__ h, const float* __restrict__ W,
    float* __restrict__ wh, unsigned short* __restrict__ whbf_t) {
    const int lane = threadIdx.x & 31;
    const int wid  = (blockIdx.x * blockDim.x + threadIdx.x) >> 5;
    const int ti = wid >> 5;            // 256 row tiles
    const int tj = wid & 31;            // 32 col tiles (HF/16)
    const int il = lane & 15;
    const int kg = lane >> 4;
    const int rowA = ti * 16 + il;
    const int colB = tj * 16 + il;

    v8f acc = {};
    #pragma unroll
    for (int k0 = 0; k0 < IN_FEAT; k0 += 32) {
        const int kb = k0 + kg * 16;    // A fragment: elems = 16 contiguous K
        const float4 av0 = *(const float4*)(h + rowA * IN_FEAT + kb);
        const float4 av1 = *(const float4*)(h + rowA * IN_FEAT + kb + 4);
        const float4 av2 = *(const float4*)(h + rowA * IN_FEAT + kb + 8);
        const float4 av3 = *(const float4*)(h + rowA * IN_FEAT + kb + 12);
        FragBF16 af;
        af.p[0] = pack2bf(av0.x, av0.y);
        af.p[1] = pack2bf(av0.z, av0.w);
        af.p[2] = pack2bf(av1.x, av1.y);
        af.p[3] = pack2bf(av1.z, av1.w);
        af.p[4] = pack2bf(av2.x, av2.y);
        af.p[5] = pack2bf(av2.z, av2.w);
        af.p[6] = pack2bf(av3.x, av3.y);
        af.p[7] = pack2bf(av3.z, av3.w);

        const int kc = k0 + kg * 16;    // B fragment: lane=col, elems = 16 contiguous K
        FragBF16 bf;
        #pragma unroll
        for (int e = 0; e < 8; ++e)
            bf.p[e] = pack2bf(W[(kc + 2 * e) * HF + colB],
                              W[(kc + 2 * e + 1) * HF + colB]);

        acc = __builtin_amdgcn_wmma_f32_16x16x32_bf16(
            false, af.v, false, bf.v, (short)0, acc, false, false);
    }

    const int m0 = ti * 16 + kg * 8;    // C layout: VGPR r -> row (kg*8 + r), col = il
    const int head = colB >> 6, f = colB & 63;
    #pragma unroll
    for (int r = 0; r < 8; ++r) {
        const float v = acc[r];
        wh[(m0 + r) * HF + colB] = v;
        whbf_t[head * (FDIM * N_NODES) + f * N_NODES + (m0 + r)] = f2bfbits(v);
    }
}

// ---------------------------------------------------------------------------
// K2: attn_1[n,h] = Wh[n,h,:].a1[h], attn_2 stored transposed [h][n]
// ---------------------------------------------------------------------------
__global__ __launch_bounds__(256) void k_attn_vec(
    const float* __restrict__ wh, const float* __restrict__ a,
    float* __restrict__ a1Out, float* __restrict__ a2tOut) {
    const int t = blockIdx.x * blockDim.x + threadIdx.x;
    const int n = t >> 3, hd = t & 7;
    const float* whr = wh + n * HF + hd * FDIM;
    const float* ar  = a + hd * (2 * FDIM);
    float s1 = 0.f, s2 = 0.f;
    #pragma unroll 8
    for (int f = 0; f < FDIM; ++f) {
        const float v = whr[f];
        s1 += v * ar[f];
        s2 += v * ar[FDIM + f];
    }
    a1Out[n * NHEAD + hd]   = s1;
    a2tOut[hd * N_NODES + n] = s2;
}

// ---------------------------------------------------------------------------
// K3: per-row masked softmax stats (online max / sum-exp), all 8 heads at once.
// One block per row i; one adj read feeds all heads.
// ---------------------------------------------------------------------------
__global__ __launch_bounds__(256) void k_softmax_stats(
    const int* __restrict__ adj, const float* __restrict__ a1,
    const float* __restrict__ a2t,
    float* __restrict__ mOut, float* __restrict__ lOut) {
    __shared__ float smM[NHEAD][256];
    __shared__ float smL[NHEAD][256];
    const int i = blockIdx.x;
    const int tid = threadIdx.x;

    float s1[NHEAD], mt[NHEAD], lt[NHEAD];
    #pragma unroll
    for (int hd = 0; hd < NHEAD; ++hd) {
        s1[hd] = a1[i * NHEAD + hd];
        mt[hd] = -3.0e38f;
        lt[hd] = 0.f;
    }

    const int* adjRow = adj + (long)i * N_NODES;
    for (int j = tid; j < N_NODES; j += 256) {
        if (adjRow[j] > 0) {
            #pragma unroll
            for (int hd = 0; hd < NHEAD; ++hd) {
                float x = s1[hd] + a2t[hd * N_NODES + j];
                x = (x > 0.f) ? x : ALPHA * x;
                if (x > mt[hd]) {
                    lt[hd] = lt[hd] * __expf(mt[hd] - x) + 1.f;
                    mt[hd] = x;
                } else {
                    lt[hd] += __expf(x - mt[hd]);
                }
            }
        }
    }
    #pragma unroll
    for (int hd = 0; hd < NHEAD; ++hd) { smM[hd][tid] = mt[hd]; smL[hd][tid] = lt[hd]; }
    __syncthreads();
    if (tid < NHEAD) {
        float M = -3.0e38f;
        for (int t = 0; t < 256; ++t) M = fmaxf(M, smM[tid][t]);
        float L = 0.f;
        for (int t = 0; t < 256; ++t) L += smL[tid][t] * __expf(smM[tid][t] - M);
        mOut[i * NHEAD + tid] = M;
        lOut[i * NHEAD + tid] = (L > 0.f) ? L : 1.f;
    }
}

// ---------------------------------------------------------------------------
// K4: h'[i,h,:] = sum_j P[i,j,h] * Wh[j,h,:]   (the 17.2 GFLOP pass)
// Block = one 16-row i-tile, 8 waves = 8 heads. Per 32-wide j chunk:
// build A fragment (probabilities) in registers, 4 bf16 WMMAs for F=64.
// ---------------------------------------------------------------------------
__global__ __launch_bounds__(256) void k_aggregate(
    const int* __restrict__ adj, const unsigned short* __restrict__ whbf_t,
    const float* __restrict__ a1, const float* __restrict__ a2t,
    const float* __restrict__ mIn, const float* __restrict__ lIn,
    const float* __restrict__ bias, float* __restrict__ out) {
    const int lane = threadIdx.x & 31;
    const int head = threadIdx.x >> 5;
    const int it   = blockIdx.x;
    const int il   = lane & 15;
    const int kg   = lane >> 4;
    const int rowA = it * 16 + il;

    const float s1    = a1[rowA * NHEAD + head];
    const float mi    = mIn[rowA * NHEAD + head];
    const float inv_l = 1.0f / lIn[rowA * NHEAD + head];

    const int* adjRow = adj + (long)rowA * N_NODES;
    const float* a2h  = a2t + head * N_NODES;
    const unsigned short* bbase = whbf_t + head * (FDIM * N_NODES);

    v8f acc[4] = {v8f{}, v8f{}, v8f{}, v8f{}};

    for (int j0 = 0; j0 < N_NODES; j0 += 32) {
        __builtin_prefetch(adjRow + j0 + 256, 0, 0);   // global_prefetch_b8
        const int kb = j0 + kg * 8;                    // A-frag K pattern
        const int4  q0 = *(const int4*)(adjRow + kb);
        const int4  q1 = *(const int4*)(adjRow + kb + 4);
        const int4  q2 = *(const int4*)(adjRow + kb + 16);
        const int4  q3 = *(const int4*)(adjRow + kb + 20);
        const float4 e0 = *(const float4*)(a2h + kb);
        const float4 e1 = *(const float4*)(a2h + kb + 4);
        const float4 e2 = *(const float4*)(a2h + kb + 16);
        const float4 e3 = *(const float4*)(a2h + kb + 20);

        FragBF16 af;
        af.p[0] = pack2bf(gat_p(q0.x, s1, e0.x, mi, inv_l),
                          gat_p(q0.y, s1, e0.y, mi, inv_l));
        af.p[1] = pack2bf(gat_p(q0.z, s1, e0.z, mi, inv_l),
                          gat_p(q0.w, s1, e0.w, mi, inv_l));
        af.p[2] = pack2bf(gat_p(q1.x, s1, e1.x, mi, inv_l),
                          gat_p(q1.y, s1, e1.y, mi, inv_l));
        af.p[3] = pack2bf(gat_p(q1.z, s1, e1.z, mi, inv_l),
                          gat_p(q1.w, s1, e1.w, mi, inv_l));
        af.p[4] = pack2bf(gat_p(q2.x, s1, e2.x, mi, inv_l),
                          gat_p(q2.y, s1, e2.y, mi, inv_l));
        af.p[5] = pack2bf(gat_p(q2.z, s1, e2.z, mi, inv_l),
                          gat_p(q2.w, s1, e2.w, mi, inv_l));
        af.p[6] = pack2bf(gat_p(q3.x, s1, e3.x, mi, inv_l),
                          gat_p(q3.y, s1, e3.y, mi, inv_l));
        af.p[7] = pack2bf(gat_p(q3.z, s1, e3.z, mi, inv_l),
                          gat_p(q3.w, s1, e3.w, mi, inv_l));

        const int kc = j0 + kg * 8;                    // B-frag: same K pattern as A
        #pragma unroll
        for (int t = 0; t < 4; ++t) {
            const unsigned short* bp = bbase + (t * 16 + il) * N_NODES + kc;
            const v8bf lo = *(const v8bf*)(bp);
            const v8bf hi = *(const v8bf*)(bp + 16);
            const v16bf bfv = __builtin_shufflevector(
                lo, hi, 0, 1, 2, 3, 4, 5, 6, 7, 8, 9, 10, 11, 12, 13, 14, 15);
            acc[t] = __builtin_amdgcn_wmma_f32_16x16x32_bf16(
                false, af.v, false, bfv, (short)0, acc[t], false, false);
        }
    }

    const int m0 = it * 16 + kg * 8;
    #pragma unroll
    for (int t = 0; t < 4; ++t) {
        const int col = head * FDIM + t * 16 + il;
        const float b = bias[col];
        #pragma unroll
        for (int r = 0; r < 8; ++r)
            out[(m0 + r) * HF + col] = acc[t][r] + b;
    }
}

// ---------------------------------------------------------------------------
extern "C" void kernel_launch(void* const* d_in, const int* in_sizes, int n_in,
                              void* d_out, int out_size, void* d_ws, size_t ws_size,
                              hipStream_t stream) {
    const float* h    = (const float*)d_in[0];   // (4096, 256)
    const int*   adj  = (const int*)d_in[1];     // (4096, 4096)
    const float* W    = (const float*)d_in[2];   // (256, 512)
    const float* a    = (const float*)d_in[3];   // (8, 128)
    const float* bias = (const float*)d_in[4];   // (512,)
    float* out = (float*)d_out;                  // (4096, 512)

    char* ws = (char*)d_ws;
    float*          wh   = (float*)(ws + 0);                  // 8 MB  f32 Wh
    unsigned short* whbf = (unsigned short*)(ws + 8388608);   // 4 MB  bf16 Wh [h][f][j]
    float*          a1b  = (float*)(ws + 12582912);           // 128 KB attn_1 [n][h]
    float*          a2t  = (float*)(ws + 12713984);           // 128 KB attn_2 [h][n]
    float*          mB   = (float*)(ws + 12845056);           // 128 KB row max
    float*          lB   = (float*)(ws + 12976128);           // 128 KB row sum-exp

    // 8192 waves (256 row tiles x 32 col tiles), 8 waves per block
    k_gemm_wh<<<1024, 256, 0, stream>>>(h, W, wh, whbf);
    k_attn_vec<<<(N_NODES * NHEAD) / 256, 256, 0, stream>>>(wh, a, a1b, a2t);
    k_softmax_stats<<<N_NODES, 256, 0, stream>>>(adj, a1b, a2t, mB, lB);
    k_aggregate<<<N_NODES / 16, 256, 0, stream>>>(adj, whbf, a1b, a2t, mB, lB, bias, out);
}
